// DotAttention_5600637354818
// MI455X (gfx1250) — matrix operand order repaired and checked by reading, not compiled, not run on hardware
//
#include <hip/hip_runtime.h>
#include <math.h>

// ---------------- CDNA5 WMMA types ----------------
typedef __attribute__((ext_vector_type(16))) __bf16 v16bf;
typedef __attribute__((ext_vector_type(8)))  float  v8f;

// ---------------- problem constants ----------------
constexpr int BATCH = 32;
constexpr int SEQLEN = 2048;
constexpr int CH = 128;          // head dim
constexpr int KT = 32;           // keys per inner tile
constexpr int WAVES = 4;         // waves per block
constexpr int ROWS_PER_WAVE = 16;
constexpr int ROWS_PER_BLOCK = WAVES * ROWS_PER_WAVE;  // 64

// padded LDS strides (bank-conflict avoidance, keep 16B alignment)
constexpr int SKP = 136;   // sK  row stride in bf16  (272 B/row)
constexpr int VKP = 40;    // sKT row stride in bf16  (80 B/row)
constexpr int SSP = 36;    // sS  row stride in f32   (144 B/row)

union BF16x16 { v16bf v; __bf16 h[16]; unsigned short u[16]; };
union BF16x2  { unsigned int u; __bf16 h[2]; };
union BF16x4  { unsigned long long u; __bf16 h[4]; };

__global__ __launch_bounds__(WAVES * 32)
void flash_attn_bf16(const float* __restrict__ seq, float* __restrict__ out) {
    const int tid  = threadIdx.x;
    const int wave = tid >> 5;
    const int lane = tid & 31;
    const int ln16 = lane & 15;   // column / row index within 16-group
    const int half = lane >> 4;   // lane half selects K sub-range in fragments

    const int blk     = blockIdx.x;
    const int b       = blk / (SEQLEN / ROWS_PER_BLOCK);
    const int rowBase = (blk % (SEQLEN / ROWS_PER_BLOCK)) * ROWS_PER_BLOCK
                      + wave * ROWS_PER_WAVE;

    const float* base = seq + (size_t)b * SEQLEN * CH;

    // LDS: double-buffered key tile in two bf16 layouts + per-wave f32 scratch
    __shared__ alignas(16) unsigned short sK [2][KT * SKP];      // [key][ch]  2x 8704 B
    __shared__ alignas(16) unsigned short sKT[2][CH * VKP];      // [ch][key]  2x10240 B
    __shared__ alignas(16) float          sS [WAVES * 16 * SSP]; // S^T scratch  9216 B
    __shared__ alignas(16) float          sStat[WAVES * 16];     //               256 B

    // cooperative staging of one 32-key tile (fp32 -> bf16, two layouts)
    auto stage = [&](int kb, int buf) {
        unsigned short* dK  = &sK [buf][0];
        unsigned short* dKT = &sKT[buf][0];
        #pragma unroll
        for (int i = 0; i < 4; ++i) {
            const int idx = tid + i * 128;        // 0..511 : 16 key-pairs x 32 ch-groups
            const int c4  = (idx >> 4) * 4;       // channel group base
            const int r   = (idx & 15) * 2;       // even key of the pair
            const float4 va = *(const float4*)(base + (size_t)(kb + r)     * CH + c4);
            const float4 vb = *(const float4*)(base + (size_t)(kb + r + 1) * CH + c4);
            BF16x4 ra, rb;
            ra.h[0]=(__bf16)va.x; ra.h[1]=(__bf16)va.y; ra.h[2]=(__bf16)va.z; ra.h[3]=(__bf16)va.w;
            rb.h[0]=(__bf16)vb.x; rb.h[1]=(__bf16)vb.y; rb.h[2]=(__bf16)vb.z; rb.h[3]=(__bf16)vb.w;
            *(unsigned long long*)&dK[(r)     * SKP + c4] = ra.u;
            *(unsigned long long*)&dK[(r + 1) * SKP + c4] = rb.u;
            #pragma unroll
            for (int j = 0; j < 4; ++j) {
                BF16x2 p; p.h[0] = ra.h[j]; p.h[1] = rb.h[j];
                *(unsigned int*)&dKT[(c4 + j) * VKP + r] = p.u;
            }
        }
    };

    // ---- load Q tile (16 x 128) as 4 bf16 A-fragments (K=32 each) ----
    v16bf qf[4];
    {
        const int qrow = rowBase + ln16;
        #pragma unroll
        for (int ci = 0; ci < 4; ++ci) {
            const int c0 = ci * 32;
            const float4 a0 = *(const float4*)(base + (size_t)qrow * CH + c0 + half * 8);
            const float4 a1 = *(const float4*)(base + (size_t)qrow * CH + c0 + half * 8 + 4);
            const float4 a2 = *(const float4*)(base + (size_t)qrow * CH + c0 + 16 + half * 8);
            const float4 a3 = *(const float4*)(base + (size_t)qrow * CH + c0 + 16 + half * 8 + 4);
            BF16x16 a;
            a.h[0]=(__bf16)a0.x; a.h[1]=(__bf16)a0.y; a.h[2]=(__bf16)a0.z; a.h[3]=(__bf16)a0.w;
            a.h[4]=(__bf16)a1.x; a.h[5]=(__bf16)a1.y; a.h[6]=(__bf16)a1.z; a.h[7]=(__bf16)a1.w;
            a.h[8]=(__bf16)a2.x; a.h[9]=(__bf16)a2.y; a.h[10]=(__bf16)a2.z; a.h[11]=(__bf16)a2.w;
            a.h[12]=(__bf16)a3.x; a.h[13]=(__bf16)a3.y; a.h[14]=(__bf16)a3.z; a.h[15]=(__bf16)a3.w;
            qf[ci] = a.v;
        }
    }

    // constant B fragment of ones (row-sum via WMMA)
    BF16x16 onesF;
    #pragma unroll
    for (int j = 0; j < 16; ++j) onesF.h[j] = (__bf16)1.0f;

    // ---- accumulators ----
    v8f o[8];
    #pragma unroll
    for (int t = 0; t < 8; ++t) o[t] = v8f{};
    v8f lacc = v8f{};                 // row sums, C-layout (all columns identical)
    float mrun = -INFINITY;           // running max for row ln16 (lane layout)

    float* pSS = sS + wave * 16 * SSP;

    constexpr int NITER = SEQLEN / KT;
    stage(0, 0);                      // prologue: stage tile 0 into buffer 0

    for (int it = 0; it < NITER; ++it) {
        const int buf = it & 1;
        __syncthreads();              // staged writes of `buf` visible; buf^1 free
        if (it + 1 < NITER)
            stage((it + 1) * KT, buf ^ 1);   // overlap next tile with compute

        const unsigned short* cK  = &sK [buf][0];
        const unsigned short* cKT = &sKT[buf][0];

        // ---- S = Q * K^T : two 16x16 tiles (keys 0-15, 16-31) ----
        v8f s0 = v8f{}, s1 = v8f{};
        #pragma unroll
        for (int ci = 0; ci < 4; ++ci) {
            const int c0 = ci * 32;
            BF16x16 b0u, b1u;
            *(uint4*)&b0u.u[0] = *(const uint4*)&cK[(ln16)      * SKP + c0 + half * 8];
            *(uint4*)&b0u.u[8] = *(const uint4*)&cK[(ln16)      * SKP + c0 + 16 + half * 8];
            *(uint4*)&b1u.u[0] = *(const uint4*)&cK[(16 + ln16) * SKP + c0 + half * 8];
            *(uint4*)&b1u.u[8] = *(const uint4*)&cK[(16 + ln16) * SKP + c0 + 16 + half * 8];
            v16bf b0v = b0u.v;
            v16bf b1v = b1u.v;
            s0 = __builtin_amdgcn_wmma_f32_16x16x32_bf16(
                     false, qf[ci], false, b0v, (short)0, s0, false, false);
            s1 = __builtin_amdgcn_wmma_f32_16x16x32_bf16(
                     false, qf[ci], false, b1v, (short)0, s1, false, false);
        }

        // ---- transpose S through per-wave LDS: [row][key] f32 ----
        #pragma unroll
        for (int r = 0; r < 8; ++r) {
            pSS[(half * 8 + r) * SSP + ln16]      = s0[r];
            pSS[(half * 8 + r) * SSP + 16 + ln16] = s1[r];
        }
        asm volatile("s_wait_dscnt 0" ::: "memory");
        const float4 x0 = *(const float4*)&pSS[ln16 * SSP + half * 8];
        const float4 x1 = *(const float4*)&pSS[ln16 * SSP + half * 8 + 4];
        const float4 x2 = *(const float4*)&pSS[ln16 * SSP + 16 + half * 8];
        const float4 x3 = *(const float4*)&pSS[ln16 * SSP + 16 + half * 8 + 4];

        // ---- row max: in-lane max + one cross-half shuffle ----
        float lm = fmaxf(fmaxf(fmaxf(x0.x, x0.y), fmaxf(x0.z, x0.w)),
                         fmaxf(fmaxf(x1.x, x1.y), fmaxf(x1.z, x1.w)));
        lm = fmaxf(lm, fmaxf(fmaxf(fmaxf(x2.x, x2.y), fmaxf(x2.z, x2.w)),
                             fmaxf(fmaxf(x3.x, x3.y), fmaxf(x3.z, x3.w))));
        lm = fmaxf(lm, __shfl_xor(lm, 16, 32));

        const bool  grew = lm > mrun;
        const float mnew = grew ? lm : mrun;

        // ---- fast path: rescale only if some row's max actually grew ----
        if (__ballot(grew)) {
            const float scl = __expf(mrun - mnew);
            sStat[wave * 16 + ln16] = scl;            // lane layout -> C layout
            asm volatile("s_wait_dscnt 0" ::: "memory");
            const float4 sc0 = *(const float4*)&sStat[wave * 16 + half * 8];
            const float4 sc1 = *(const float4*)&sStat[wave * 16 + half * 8 + 4];
            const float scl8[8] = {sc0.x, sc0.y, sc0.z, sc0.w, sc1.x, sc1.y, sc1.z, sc1.w};
            #pragma unroll
            for (int t = 0; t < 8; ++t)
                #pragma unroll
                for (int r = 0; r < 8; ++r)
                    o[t][r] *= scl8[r];
            #pragma unroll
            for (int r = 0; r < 8; ++r) lacc[r] *= scl8[r];
            mrun = mnew;
        }

        // ---- P = exp(S - m), straight into the bf16 A-fragment ----
        BF16x16 pa;
        pa.h[0]  = (__bf16)__expf(x0.x - mnew); pa.h[1]  = (__bf16)__expf(x0.y - mnew);
        pa.h[2]  = (__bf16)__expf(x0.z - mnew); pa.h[3]  = (__bf16)__expf(x0.w - mnew);
        pa.h[4]  = (__bf16)__expf(x1.x - mnew); pa.h[5]  = (__bf16)__expf(x1.y - mnew);
        pa.h[6]  = (__bf16)__expf(x1.z - mnew); pa.h[7]  = (__bf16)__expf(x1.w - mnew);
        pa.h[8]  = (__bf16)__expf(x2.x - mnew); pa.h[9]  = (__bf16)__expf(x2.y - mnew);
        pa.h[10] = (__bf16)__expf(x2.z - mnew); pa.h[11] = (__bf16)__expf(x2.w - mnew);
        pa.h[12] = (__bf16)__expf(x3.x - mnew); pa.h[13] = (__bf16)__expf(x3.y - mnew);
        pa.h[14] = (__bf16)__expf(x3.z - mnew); pa.h[15] = (__bf16)__expf(x3.w - mnew);

        // ---- row sums via WMMA with B = ones ----
        lacc = __builtin_amdgcn_wmma_f32_16x16x32_bf16(
                   false, pa.v, false, onesF.v, (short)0, lacc, false, false);

        // ---- O += P * V over 8 channel tiles (transposed staged tile) ----
        #pragma unroll
        for (int nc = 0; nc < 8; ++nc) {
            BF16x16 bu;
            *(uint4*)&bu.u[0] = *(const uint4*)&cKT[(nc * 16 + ln16) * VKP + half * 8];
            *(uint4*)&bu.u[8] = *(const uint4*)&cKT[(nc * 16 + ln16) * VKP + 16 + half * 8];
            v16bf bv = bu.v;
            o[nc] = __builtin_amdgcn_wmma_f32_16x16x32_bf16(
                        false, pa.v, false, bv, (short)0, o[nc], false, false);
        }
    }

    // ---- epilogue: O /= l (lacc already in C-layout), write out ----
    float inv[8];
    #pragma unroll
    for (int r = 0; r < 8; ++r) inv[r] = 1.0f / lacc[r];

    float* op = out + (size_t)b * SEQLEN * CH;
    #pragma unroll
    for (int nc = 0; nc < 8; ++nc) {
        #pragma unroll
        for (int r = 0; r < 8; ++r) {
            const int row = rowBase + r + half * 8;
            const int ch  = nc * 16 + ln16;
            op[(size_t)row * CH + ch] = o[nc][r] * inv[r];
        }
    }
}

extern "C" void kernel_launch(void* const* d_in, const int* in_sizes, int n_in,
                              void* d_out, int out_size, void* d_ws, size_t ws_size,
                              hipStream_t stream) {
    const float* seq = (const float*)d_in[0];
    float* out = (float*)d_out;
    const int nblk = BATCH * (SEQLEN / ROWS_PER_BLOCK);   // 1024
    flash_attn_bf16<<<nblk, WAVES * 32, 0, stream>>>(seq, out);
}
